// DeepFM_82471962018408
// MI455X (gfx1250) — compile-verified, matrix operand order checked by reading, not run on hardware
//
#include <hip/hip_runtime.h>
#include <hip/hip_bf16.h>

typedef __attribute__((ext_vector_type(16))) _Float16 v16h;
typedef __attribute__((ext_vector_type(8)))  _Float16 v8h;
typedef __attribute__((ext_vector_type(8)))  float    v8f;

#define WMMA_F16(a, b, c) \
    __builtin_amdgcn_wmma_f32_16x16x32_f16(false, (a), false, (b), (short)0, (c), false, false)

namespace {
constexpr int Bc    = 16384;
constexpr int F     = 26;
constexpr int Dd    = 13;
constexpr int E     = 16;
constexpr int V1    = 100001;
constexpr int H1    = 128;
constexpr int H2    = 64;
constexpr int INDIM = F * E + Dd;   // 429
constexpr int KP1   = 448;          // 14 * 32  (K padded for GEMM1)
constexpr float EPS = 1e-5f;

// workspace layout (bytes, all 16B-aligned)
constexpr size_t OFF_W1H = 0;                          // 128*448 f16 = 114688
constexpr size_t OFF_W2H = OFF_W1H + 128 * 448 * 2;    // 64*128 f16  = 16384
constexpr size_t OFF_S0  = OFF_W2H + 64 * 128 * 2;     // 432 f32
constexpr size_t OFF_T0  = OFF_S0  + 432 * 4;          // 432 f32
constexpr size_t OFF_SH1 = OFF_T0  + 432 * 4;          // 128 f32
constexpr size_t OFF_SH2 = OFF_SH1 + 128 * 4;          // 64 f32
constexpr size_t WS_NEED = OFF_SH2 + 64 * 4;           // 135296 bytes
}

// ---------------------------------------------------------------------------
// Prep kernel: BN-fold weights into packed f16, precompute BN scale/shift vecs
// ---------------------------------------------------------------------------
__global__ __launch_bounds__(256) void deepfm_prep_kernel(
    const float* __restrict__ bn0g, const float* __restrict__ bn0b,
    const float* __restrict__ bn0m, const float* __restrict__ bn0v,
    const float* __restrict__ W1,   const float* __restrict__ b1,
    const float* __restrict__ bn1g, const float* __restrict__ bn1b,
    const float* __restrict__ bn1m, const float* __restrict__ bn1v,
    const float* __restrict__ W2,   const float* __restrict__ b2,
    const float* __restrict__ bn2g, const float* __restrict__ bn2b,
    const float* __restrict__ bn2m, const float* __restrict__ bn2v,
    _Float16* __restrict__ W1h, _Float16* __restrict__ W2h,
    float* __restrict__ s0, float* __restrict__ t0,
    float* __restrict__ sh1, float* __restrict__ sh2)
{
    const int stride = gridDim.x * blockDim.x;
    const int tid0   = blockIdx.x * blockDim.x + threadIdx.x;

    // W1h[n][k] = W1[n][k] * scale1[n], zero-padded to K=448
    for (int i = tid0; i < H1 * KP1; i += stride) {
        const int n = i / KP1, k = i - n * KP1;
        const float sc = bn1g[n] * rsqrtf(bn1v[n] + EPS);
        W1h[i] = (_Float16)((k < INDIM) ? W1[(size_t)n * INDIM + k] * sc : 0.f);
    }
    // W2h[n][k] = W2[n][k] * scale2[n]
    for (int i = tid0; i < H2 * H1; i += stride) {
        const int n = i / H1;
        const float sc = bn2g[n] * rsqrtf(bn2v[n] + EPS);
        W2h[i] = (_Float16)(W2[i] * sc);
    }
    // BN0 scale/shift
    for (int i = tid0; i < INDIM; i += stride) {
        const float sc = bn0g[i] * rsqrtf(bn0v[i] + EPS);
        s0[i] = sc;
        t0[i] = bn0b[i] - bn0m[i] * sc;
    }
    // BN1/BN2 output shifts (linear bias folded in)
    for (int i = tid0; i < H1; i += stride) {
        const float sc = bn1g[i] * rsqrtf(bn1v[i] + EPS);
        sh1[i] = (b1[i] - bn1m[i]) * sc + bn1b[i];
    }
    for (int i = tid0; i < H2; i += stride) {
        const float sc = bn2g[i] * rsqrtf(bn2v[i] + EPS);
        sh2[i] = (b2[i] - bn2m[i]) * sc + bn2b[i];
    }
}

// ---------------------------------------------------------------------------
// Fused DeepFM kernel: 1 workgroup (8 waves) = 16 batch rows
// ---------------------------------------------------------------------------
template<bool PACKED>
__global__ __launch_bounds__(256) void deepfm_fused_kernel(
    const int*   __restrict__ sparse,   // [B,F]
    const float* __restrict__ dense,    // [B,D]
    const float* __restrict__ embt,     // [F,V1,E]
    const float* __restrict__ lint,     // [F,V1]
    const float* __restrict__ ldw,      // [D]
    const float* __restrict__ ldb,      // [1]
    const float* __restrict__ bn0g, const float* __restrict__ bn0b,
    const float* __restrict__ bn0m, const float* __restrict__ bn0v,
    const float* __restrict__ W1,   const float* __restrict__ b1,
    const float* __restrict__ bn1g, const float* __restrict__ bn1b,
    const float* __restrict__ bn1m, const float* __restrict__ bn1v,
    const float* __restrict__ W2,   const float* __restrict__ b2,
    const float* __restrict__ bn2g, const float* __restrict__ bn2b,
    const float* __restrict__ bn2m, const float* __restrict__ bn2v,
    const float* __restrict__ Wo,   const float* __restrict__ bo,
    const float* __restrict__ gbias,
    const _Float16* __restrict__ W1h, const _Float16* __restrict__ W2h,
    const float* __restrict__ s0v, const float* __restrict__ t0v,
    const float* __restrict__ sh1v, const float* __restrict__ sh2v,
    float*       __restrict__ out)      // [B]
{
    __shared__ _Float16 Xs [16 * KP1];   // BN0-normalized deep input tile (f16)
    __shared__ _Float16 A1s[16 * H1];    // hidden-1 activations (f16)
    __shared__ _Float16 A2s[16 * H2];    // hidden-2 activations (f16)
    __shared__ float    partial[16];     // gbias + linear + FM per row

    const int t = threadIdx.x;

    // ---------------- Phase 1: gather + FM + linear + BN0 -> LDS -------------
    {
        const int row = t >> 4;          // 0..15 (local batch row)
        const int e   = t & 15;          // embedding dim
        const long grow = (long)blockIdx.x * 16 + row;
        const int* sidx = sparse + grow * F;

        float s_acc = 0.f, sq_acc = 0.f;
        #pragma unroll
        for (int f = 0; f < F; ++f) {
            const int ix = sidx[f];
            const float v = embt[(size_t)f * V1 * E + (size_t)ix * E + e];
            s_acc  += v;
            sq_acc += v * v;
            const int k = f * E + e;
            float sc, sh;
            if constexpr (PACKED) { sc = s0v[k]; sh = t0v[k]; }
            else {
                sc = bn0g[k] * rsqrtf(bn0v[k] + EPS);
                sh = bn0b[k] - bn0m[k] * sc;
            }
            Xs[row * KP1 + k] = (_Float16)(v * sc + sh);
        }

        // dense part of deep_in + linear-dense partial
        float ld_part = 0.f;
        if (e < Dd) {
            const float xd = dense[grow * Dd + e];
            ld_part = xd * ldw[e];
            const int k = F * E + e;     // 416..428
            float sc, sh;
            if constexpr (PACKED) { sc = s0v[k]; sh = t0v[k]; }
            else {
                sc = bn0g[k] * rsqrtf(bn0v[k] + EPS);
                sh = bn0b[k] - bn0m[k] * sc;
            }
            Xs[row * KP1 + k] = (_Float16)(xd * sc + sh);
        }
        // zero the K padding 429..447
        Xs[row * KP1 + 432 + e] = (_Float16)0.f;
        if (e < 3) Xs[row * KP1 + 429 + e] = (_Float16)0.f;

        // linear-sparse partial: features e and e+16
        float lp = 0.f;
        {
            const int ix = sidx[e];
            lp += lint[(size_t)e * V1 + ix];
        }
        if (e + 16 < F) {
            const int ix = sidx[e + 16];
            lp += lint[(size_t)(e + 16) * V1 + ix];
        }

        float fm_part = s_acc * s_acc - sq_acc;

        // reduce across the 16 lanes of this row (stays inside 16-lane half)
        #pragma unroll
        for (int m = 8; m >= 1; m >>= 1) {
            fm_part += __shfl_xor(fm_part, m, 32);
            lp      += __shfl_xor(lp,      m, 32);
            ld_part += __shfl_xor(ld_part, m, 32);
        }
        if (e == 0)
            partial[row] = gbias[0] + lp + ld_part + ldb[0] + 0.5f * fm_part;
    }
    __syncthreads();

    // common lane geometry for WMMA phases
    const int wave   = t >> 5;
    const int lane   = t & 31;
    const int lhalf  = lane & 15;        // M (A frag) / N (B,C frags)
    const int hi     = lane >> 4;        // K-half selector
    const int koff   = hi * 8;
    const int rowoff = hi * 8;           // C/D: lanes 16-31 hold M = r+8

    // ---------------- Phase 2: GEMM1 (16x448 @ 448x16 per wave, 8 N tiles) ---
    {
        const int ncol = wave * 16 + lhalf;                // 0..127
        float sh1;
        if constexpr (PACKED) sh1 = sh1v[ncol];
        else {
            const float sc = bn1g[ncol] * rsqrtf(bn1v[ncol] + EPS);
            sh1 = (b1[ncol] - bn1m[ncol]) * sc + bn1b[ncol];
        }

        v8f acc = {};
        for (int kk = 0; kk < KP1 / 32; ++kk) {
            const int kbase = kk * 32;
            // A fragment: ISA 16-bit 16x32 layout, two contiguous 16B LDS reads
            const v8h alo = *(const v8h*)&Xs[lhalf * KP1 + kbase + koff];
            const v8h ahi = *(const v8h*)&Xs[lhalf * KP1 + kbase + 16 + koff];
            const v16h a = __builtin_shufflevector(alo, ahi, 0,1,2,3,4,5,6,7,
                                                             8,9,10,11,12,13,14,15);
            v16h bf;
            if constexpr (PACKED) {
                // packed f16, scale pre-folded: two 16B global loads
                const v8h blo = *(const v8h*)&W1h[(size_t)ncol * KP1 + kbase + koff];
                const v8h bhi = *(const v8h*)&W1h[(size_t)ncol * KP1 + kbase + 16 + koff];
                bf = __builtin_shufflevector(blo, bhi, 0,1,2,3,4,5,6,7,
                                                       8,9,10,11,12,13,14,15);
            } else {
                const float sc = bn1g[ncol] * rsqrtf(bn1v[ncol] + EPS);
                #pragma unroll
                for (int h = 0; h < 16; ++h) {
                    const int kl = ((h < 8) ? h : h + 8) + koff;
                    const int kg = kbase + kl;
                    bf[h] = (_Float16)((kg < INDIM)
                                       ? W1[(size_t)ncol * INDIM + kg] * sc : 0.f);
                }
            }
            acc = WMMA_F16(a, bf, acc);
        }
        #pragma unroll
        for (int r = 0; r < 8; ++r) {
            float v = acc[r] + sh1;
            v = fmaxf(v, 0.f);                              // ReLU
            A1s[(r + rowoff) * H1 + ncol] = (_Float16)v;
        }
    }
    __syncthreads();

    // ---------------- Phase 3: GEMM2 (16x128 @ 128x16, 4 N tiles) ------------
    if (wave < 4) {
        const int ncol = wave * 16 + lhalf;                 // 0..63
        float sh2;
        if constexpr (PACKED) sh2 = sh2v[ncol];
        else {
            const float sc = bn2g[ncol] * rsqrtf(bn2v[ncol] + EPS);
            sh2 = (b2[ncol] - bn2m[ncol]) * sc + bn2b[ncol];
        }

        v8f acc = {};
        #pragma unroll
        for (int kk = 0; kk < H1 / 32; ++kk) {
            const int kbase = kk * 32;
            const v8h alo = *(const v8h*)&A1s[lhalf * H1 + kbase + koff];
            const v8h ahi = *(const v8h*)&A1s[lhalf * H1 + kbase + 16 + koff];
            const v16h a = __builtin_shufflevector(alo, ahi, 0,1,2,3,4,5,6,7,
                                                             8,9,10,11,12,13,14,15);
            v16h bf;
            if constexpr (PACKED) {
                const v8h blo = *(const v8h*)&W2h[(size_t)ncol * H1 + kbase + koff];
                const v8h bhi = *(const v8h*)&W2h[(size_t)ncol * H1 + kbase + 16 + koff];
                bf = __builtin_shufflevector(blo, bhi, 0,1,2,3,4,5,6,7,
                                                       8,9,10,11,12,13,14,15);
            } else {
                const float sc = bn2g[ncol] * rsqrtf(bn2v[ncol] + EPS);
                #pragma unroll
                for (int h = 0; h < 16; ++h) {
                    const int kl = ((h < 8) ? h : h + 8) + koff;
                    bf[h] = (_Float16)(W2[(size_t)ncol * H1 + kbase + kl] * sc);
                }
            }
            acc = WMMA_F16(a, bf, acc);
        }
        #pragma unroll
        for (int r = 0; r < 8; ++r) {
            float v = acc[r] + sh2;
            v = fmaxf(v, 0.f);
            A2s[(r + rowoff) * H2 + ncol] = (_Float16)v;
        }
    }
    __syncthreads();

    // ---------------- Phase 4: output head -----------------------------------
    if (t < 16) {
        float dot = 0.f;
        #pragma unroll
        for (int k = 0; k < H2; ++k)
            dot += (float)A2s[t * H2 + k] * Wo[k];
        out[(long)blockIdx.x * 16 + t] = partial[t] + dot + bo[0];
    }
}

extern "C" void kernel_launch(void* const* d_in, const int* in_sizes, int n_in,
                              void* d_out, int out_size, void* d_ws, size_t ws_size,
                              hipStream_t stream) {
    (void)in_sizes; (void)n_in; (void)out_size;
    const int*   sparse = (const int*)  d_in[0];
    const float* dense  = (const float*)d_in[1];
    const float* embt   = (const float*)d_in[2];
    const float* lint   = (const float*)d_in[3];
    const float* ldw    = (const float*)d_in[4];
    const float* ldb    = (const float*)d_in[5];
    const float* bn0g   = (const float*)d_in[6];
    const float* bn0b   = (const float*)d_in[7];
    const float* bn0m   = (const float*)d_in[8];
    const float* bn0v   = (const float*)d_in[9];
    const float* W1     = (const float*)d_in[10];
    const float* b1     = (const float*)d_in[11];
    const float* bn1g   = (const float*)d_in[12];
    const float* bn1b   = (const float*)d_in[13];
    const float* bn1m   = (const float*)d_in[14];
    const float* bn1v   = (const float*)d_in[15];
    const float* W2     = (const float*)d_in[16];
    const float* b2     = (const float*)d_in[17];
    const float* bn2g   = (const float*)d_in[18];
    const float* bn2b   = (const float*)d_in[19];
    const float* bn2m   = (const float*)d_in[20];
    const float* bn2v   = (const float*)d_in[21];
    const float* Wo     = (const float*)d_in[22];
    const float* bo     = (const float*)d_in[23];
    const float* gbias  = (const float*)d_in[24];
    float*       out    = (float*)d_out;

    dim3 grid(Bc / 16), block(256);

    if (ws_size >= WS_NEED && d_ws != nullptr) {
        char* ws = (char*)d_ws;
        _Float16* W1h = (_Float16*)(ws + OFF_W1H);
        _Float16* W2h = (_Float16*)(ws + OFF_W2H);
        float*    s0  = (float*)(ws + OFF_S0);
        float*    t0  = (float*)(ws + OFF_T0);
        float*    sh1 = (float*)(ws + OFF_SH1);
        float*    sh2 = (float*)(ws + OFF_SH2);

        deepfm_prep_kernel<<<dim3(64), dim3(256), 0, stream>>>(
            bn0g, bn0b, bn0m, bn0v,
            W1, b1, bn1g, bn1b, bn1m, bn1v,
            W2, b2, bn2g, bn2b, bn2m, bn2v,
            W1h, W2h, s0, t0, sh1, sh2);

        deepfm_fused_kernel<true><<<grid, block, 0, stream>>>(
            sparse, dense, embt, lint, ldw, ldb,
            bn0g, bn0b, bn0m, bn0v,
            W1, b1, bn1g, bn1b, bn1m, bn1v,
            W2, b2, bn2g, bn2b, bn2m, bn2v,
            Wo, bo, gbias,
            W1h, W2h, s0, t0, sh1, sh2, out);
    } else {
        deepfm_fused_kernel<false><<<grid, block, 0, stream>>>(
            sparse, dense, embt, lint, ldw, ldb,
            bn0g, bn0b, bn0m, bn0v,
            W1, b1, bn1g, bn1b, bn1m, bn1v,
            W2, b2, bn2g, bn2b, bn2m, bn2v,
            Wo, bo, gbias,
            nullptr, nullptr, nullptr, nullptr, nullptr, nullptr, out);
    }
}
